// MHA_28922309771622
// MI455X (gfx1250) — matrix operand order repaired and checked by reading, compiled, not run on hardware
//
#include <hip/hip_runtime.h>
#include <hip/hip_bf16.h>

// ---------------------------------------------------------------------------
// MHA with memory (Transformer-XL style), bf16 WMMA compute / f32 accumulate.
// B=2 S=2048 MEM=512 D=1024 H=16 DH=64 Skv=2560
// ---------------------------------------------------------------------------

typedef __attribute__((ext_vector_type(16))) __bf16 v16bf;
typedef __attribute__((ext_vector_type(8)))  __bf16 v8bf;
typedef __attribute__((ext_vector_type(8)))  float  v8f;

#define WMMA_BF16(A, Bm, C) \
    __builtin_amdgcn_wmma_f32_16x16x32_bf16(false, (A), false, (Bm), (short)0, (C), false, false)

namespace cfg {
constexpr int B   = 2;
constexpr int S   = 2048;
constexpr int MEM = 512;
constexpr int D   = 1024;
constexpr int H   = 16;
constexpr int DH  = 64;
constexpr int SKV = MEM + S;  // 2560
}

// Pack 8 consecutive f32 (two float4) into bf16 vector elements [off, off+8).
__device__ __forceinline__ void cvt8(v16bf& dst, int off, float4 a, float4 b) {
    dst[off + 0] = (__bf16)a.x; dst[off + 1] = (__bf16)a.y;
    dst[off + 2] = (__bf16)a.z; dst[off + 3] = (__bf16)a.w;
    dst[off + 4] = (__bf16)b.x; dst[off + 5] = (__bf16)b.y;
    dst[off + 6] = (__bf16)b.z; dst[off + 7] = (__bf16)b.w;
}

// ---------------------------------------------------------------------------
// Kernel 1: projection GEMM  out = gather(X) @ W^T + bias   (bf16 out)
//   CONCAT=0 : rows from x [B,S,D]        (T = S)
//   CONCAT=1 : rows from concat(mems, x)  (T = SKV)
//   VT=0     : out[b][h][t][dh]   (Q / K layout)
//   VT=1     : out[b][h][dh][t]   (V transposed -> K-major PV fragments)
// One wave computes a 16x64 output tile: one A fragment shared by 4 chained
// WMMAs per 32-deep K step.
// ---------------------------------------------------------------------------
template<int T, bool CONCAT, bool VT>
__global__ __launch_bounds__(32) void proj_qkv(
    const float* __restrict__ x, const float* __restrict__ mems,
    const float* __restrict__ W, const float* __restrict__ bias,
    __bf16* __restrict__ out)
{
    using namespace cfg;
    const int lane = threadIdx.x;
    const int g  = lane >> 4;    // half-wave select
    const int ln = lane & 15;

    constexpr int tilesN = D / 64;
    const int n0   = (blockIdx.x % tilesN) * 64;
    const int row0 = (blockIdx.x / tilesN) * 16;

    const int rA = row0 + ln;
    const float* arow;
    if (CONCAT) {
        const int bI = rA / SKV;                      // compile-time divisor
        const int tI = rA - bI * SKV;
        arow = (tI < MEM) ? (mems + (size_t)(bI * MEM + tI) * D)
                          : (x    + (size_t)(bI * S + (tI - MEM)) * D);
    } else {
        arow = x + (size_t)rA * D;
    }

    const int kb = 16 * g;
    const float* wrow[4];
#pragma unroll
    for (int t = 0; t < 4; ++t)
        wrow[t] = W + (size_t)(n0 + 16 * t + ln) * D + 16 * g;

    v8f acc[4];
#pragma unroll
    for (int t = 0; t < 4; ++t) acc[t] = (v8f){};

    for (int k0 = 0; k0 < D; k0 += 32) {
        v16bf a;
        {
            const float4* p0 = (const float4*)(arow + k0 + kb);
            const float4* p1 = (const float4*)(arow + k0 + kb + 8);
            cvt8(a, 0, p0[0], p0[1]);
            cvt8(a, 8, p1[0], p1[1]);
        }
#pragma unroll
        for (int t = 0; t < 4; ++t) {
            v16bf bm;
            const float4* p = (const float4*)(wrow[t] + k0);
            cvt8(bm, 0, p[0], p[1]);
            cvt8(bm, 8, p[2], p[3]);
            acc[t] = WMMA_BF16(a, bm, acc[t]);
        }
    }

#pragma unroll
    for (int t = 0; t < 4; ++t) {
        const int nCol = n0 + 16 * t + ln;
        const float bb = bias[nCol];
        const int h  = nCol >> 6;
        const int dh = nCol & 63;
#pragma unroll
        for (int r = 0; r < 8; ++r) {
            const int row = row0 + r + 8 * g;
            const int bO  = row / T;
            const int trw = row - bO * T;
            const float v = acc[t][r] + bb;
            const size_t idx = VT
                ? ((size_t)(bO * H + h) * DH + dh) * (size_t)T + trw
                : ((size_t)(bO * H + h) * T + trw) * DH + dh;
            out[idx] = (__bf16)v;
        }
    }
}

// ---------------------------------------------------------------------------
// Kernel 2: flash attention. One wave per (b, h, 32-query block).
// Two 16-row query tiles share every K / V fragment load (halves L2 traffic,
// doubles WMMA per byte). Row sums of P are computed with one extra WMMA
// against a ones-fragment (P @ 1) instead of 32 ds_bpermute stages; only the
// running-max reduction uses lane shuffles.
//   Q,K: [B,H,*,64] bf16;  Vt: [B,H,64,SKV] bf16;  Ctx: [B,S,D] bf16.
// ---------------------------------------------------------------------------
__global__ __launch_bounds__(32) void attn_flash(
    const __bf16* __restrict__ Qb, const __bf16* __restrict__ Kb,
    const __bf16* __restrict__ Vt, const int* __restrict__ mask,
    __bf16* __restrict__ Ctx)
{
    using namespace cfg;
    __shared__ __bf16 pLds[32][40];                 // 2x 16x32 P tiles, 80B rows

    const int lane = threadIdx.x;
    const int g  = lane >> 4;
    const int ln = lane & 15;

    constexpr int qBlocks = S / 32;
    const int qblk = blockIdx.x % qBlocks;
    const int bh   = blockIdx.x / qBlocks;
    const int b    = bh / H;
    const int h    = bh % H;
    const int q0   = qblk * 32;

    // Q fragments: 2 query tiles x 2 K-chunks of DH, resident all loop long.
    v16bf aq[2][2];
#pragma unroll
    for (int q = 0; q < 2; ++q) {
        const __bf16* qrow = Qb + ((size_t)bh * S + q0 + 16 * q + ln) * DH;
#pragma unroll
        for (int kc = 0; kc < 2; ++kc) {
            v8bf lo = *(const v8bf*)(qrow + kc * 32 + 16 * g);
            v8bf hi = *(const v8bf*)(qrow + kc * 32 + 16 * g + 8);
#pragma unroll
            for (int e = 0; e < 8; ++e) { aq[q][kc][e] = lo[e]; aq[q][kc][8 + e] = hi[e]; }
        }
    }

    // Ones B-fragment for WMMA row sums (P @ 1 -> row sums broadcast to lanes).
    v16bf ones;
#pragma unroll
    for (int e = 0; e < 16; ++e) ones[e] = (__bf16)1.0f;

    float mrun[2][8], lrun[2][8];
    v8f   acc[2][4];
#pragma unroll
    for (int q = 0; q < 2; ++q) {
#pragma unroll
        for (int r = 0; r < 8; ++r) { mrun[q][r] = -1e30f; lrun[q][r] = 0.0f; }
#pragma unroll
        for (int t = 0; t < 4; ++t) acc[q][t] = (v8f){};
    }

    const __bf16* kbase = Kb + (size_t)bh * SKV * DH;
    const __bf16* vbase = Vt + (size_t)bh * DH * SKV;
    const int*    mrow  = mask + b * SKV;

    for (int kv0 = 0; kv0 < SKV; kv0 += 32) {
        // Prefetch (global_prefetch_b8): K = one 128B line per next-chunk row;
        // V lines (64 kv per line) refreshed once per 64-kv boundary, covering
        // all 64 dh rows with two prefetch ops.
        if (kv0 + 32 < SKV) {
            __builtin_prefetch(kbase + (size_t)(kv0 + 32 + lane) * DH, 0, 0);
            if ((kv0 & 32) == 0) {
                __builtin_prefetch(vbase + (size_t)lane        * SKV + kv0 + 64, 0, 0);
                __builtin_prefetch(vbase + (size_t)(lane + 32) * SKV + kv0 + 64, 0, 0);
            }
        }

        // ---- K^T fragments, shared by both query tiles ----
        v16bf bk[2][2];                             // [j = N tile][kc = K chunk]
        float pen[2];
#pragma unroll
        for (int j = 0; j < 2; ++j) {
            const int col = kv0 + 16 * j + ln;
#pragma unroll
            for (int kc = 0; kc < 2; ++kc)
                bk[j][kc] = *(const v16bf*)(kbase + (size_t)col * DH + kc * 32 + 16 * g);
            pen[j] = mrow[col] ? 0.0f : -1.0e6f;
        }

        // ---- scores: 2 query tiles x 2 kv tiles ----
        v8f st[2][2];
#pragma unroll
        for (int q = 0; q < 2; ++q) {
#pragma unroll
            for (int j = 0; j < 2; ++j) {
                v8f c = {};
#pragma unroll
                for (int kc = 0; kc < 2; ++kc)
                    c = WMMA_BF16(aq[q][kc], bk[j][kc], c);
                st[q][j] = c;
            }
        }

        // ---- online softmax: max via lane shuffles, P -> LDS as bf16 ----
        float alphaS[2][8];
#pragma unroll
        for (int q = 0; q < 2; ++q) {
#pragma unroll
            for (int r = 0; r < 8; ++r) {
                float s0 = st[q][0][r] * 0.125f + pen[0];
                float s1 = st[q][1][r] * 0.125f + pen[1];
                float mx = fmaxf(s0, s1);
#pragma unroll
                for (int off = 1; off < 16; off <<= 1)
                    mx = fmaxf(mx, __shfl_xor(mx, off, 32));
                const float nm    = fmaxf(mrun[q][r], mx);
                const float alpha = __expf(mrun[q][r] - nm);
                mrun[q][r]  = nm;
                alphaS[q][r] = alpha;
                const float p0 = __expf(s0 - nm);
                const float p1 = __expf(s1 - nm);
#pragma unroll
                for (int t = 0; t < 4; ++t) acc[q][t][r] = acc[q][t][r] * alpha;
                const int row = 16 * q + r + 8 * g;
                pLds[row][ln]      = (__bf16)p0;
                pLds[row][16 + ln] = (__bf16)p1;
            }
        }
        __syncthreads();

        // ---- re-layout P from C-fragment to A-fragment via LDS ----
        v16bf ap[2];
#pragma unroll
        for (int q = 0; q < 2; ++q) {
            v8bf lo = *(const v8bf*)&pLds[16 * q + ln][16 * g];
            v8bf hi = *(const v8bf*)&pLds[16 * q + ln][16 * g + 8];
#pragma unroll
            for (int e = 0; e < 8; ++e) { ap[q][e] = lo[e]; ap[q][8 + e] = hi[e]; }
        }
        __syncthreads();

        // ---- row sums via WMMA (P @ ones), then lrun update ----
#pragma unroll
        for (int q = 0; q < 2; ++q) {
            v8f z = {};
            v8f rs = WMMA_BF16(ap[q], ones, z);
#pragma unroll
            for (int r = 0; r < 8; ++r)
                lrun[q][r] = lrun[q][r] * alphaS[q][r] + rs[r];
        }

        // ---- O += P @ V : V fragments shared by both query tiles ----
#pragma unroll
        for (int t = 0; t < 4; ++t) {
            const int dh = 16 * t + ln;
            v16bf bv = *(const v16bf*)(vbase + (size_t)dh * SKV + kv0 + 16 * g);
#pragma unroll
            for (int q = 0; q < 2; ++q)
                acc[q][t] = WMMA_BF16(ap[q], bv, acc[q][t]);
        }
    }

    // ---- normalize and store context (merged heads, [B,S,D]) ----
#pragma unroll
    for (int q = 0; q < 2; ++q) {
#pragma unroll
        for (int r = 0; r < 8; ++r) {
            const float inv = 1.0f / lrun[q][r];
            const int row   = q0 + 16 * q + r + 8 * g;
            const size_t base = ((size_t)b * S + row) * D + h * DH;
#pragma unroll
            for (int t = 0; t < 4; ++t)
                Ctx[base + 16 * t + ln] = (__bf16)(acc[q][t][r] * inv);
        }
    }
}

// ---------------------------------------------------------------------------
// Kernel 3: output projection  out = Ctx(bf16) @ Wo^T + bo   (f32 out)
// ---------------------------------------------------------------------------
__global__ __launch_bounds__(32) void out_proj(
    const __bf16* __restrict__ Ctx, const float* __restrict__ Wo,
    const float* __restrict__ bo, float* __restrict__ out)
{
    using namespace cfg;
    const int lane = threadIdx.x;
    const int g  = lane >> 4;
    const int ln = lane & 15;

    constexpr int tilesN = D / 64;
    const int n0   = (blockIdx.x % tilesN) * 64;
    const int row0 = (blockIdx.x / tilesN) * 16;

    const __bf16* arow = Ctx + (size_t)(row0 + ln) * D;
    const int kb = 16 * g;
    const float* wrow[4];
#pragma unroll
    for (int t = 0; t < 4; ++t)
        wrow[t] = Wo + (size_t)(n0 + 16 * t + ln) * D + 16 * g;

    v8f acc[4];
#pragma unroll
    for (int t = 0; t < 4; ++t) acc[t] = (v8f){};

    for (int k0 = 0; k0 < D; k0 += 32) {
        v16bf a;
        {
            v8bf lo = *(const v8bf*)(arow + k0 + kb);
            v8bf hi = *(const v8bf*)(arow + k0 + kb + 8);
#pragma unroll
            for (int e = 0; e < 8; ++e) { a[e] = lo[e]; a[8 + e] = hi[e]; }
        }
#pragma unroll
        for (int t = 0; t < 4; ++t) {
            v16bf bm;
            const float4* p = (const float4*)(wrow[t] + k0);
            cvt8(bm, 0, p[0], p[1]);
            cvt8(bm, 8, p[2], p[3]);
            acc[t] = WMMA_BF16(a, bm, acc[t]);
        }
    }

#pragma unroll
    for (int t = 0; t < 4; ++t) {
        const int nCol = n0 + 16 * t + ln;
        const float bb = bo[nCol];
#pragma unroll
        for (int r = 0; r < 8; ++r) {
            const int row = row0 + r + 8 * g;
            out[(size_t)row * D + nCol] = acc[t][r] + bb;
        }
    }
}

// ---------------------------------------------------------------------------
// Launch
// ---------------------------------------------------------------------------
extern "C" void kernel_launch(void* const* d_in, const int* in_sizes, int n_in,
                              void* d_out, int out_size, void* d_ws, size_t ws_size,
                              hipStream_t stream) {
    using namespace cfg;
    const float* x    = (const float*)d_in[0];
    const float* mems = (const float*)d_in[1];
    // d_in[2] = freqs_cis : unused by the reference
    const int*   mask = (const int*)d_in[3];
    const float* Wq = (const float*)d_in[4];  const float* bq = (const float*)d_in[5];
    const float* Wk = (const float*)d_in[6];  const float* bk = (const float*)d_in[7];
    const float* Wv = (const float*)d_in[8];  const float* bv = (const float*)d_in[9];
    const float* Wo = (const float*)d_in[10]; const float* bo = (const float*)d_in[11];
    float* out = (float*)d_out;

    // Workspace layout (bytes): Q 8MiB | K 10MiB | Vt 10MiB | Ctx 8MiB = 36MiB
    char* ws = (char*)d_ws;
    __bf16* Qb  = (__bf16*)(ws);
    __bf16* Kb  = (__bf16*)(ws + (size_t)8  * 1024 * 1024);
    __bf16* Vt  = (__bf16*)(ws + (size_t)18 * 1024 * 1024);
    __bf16* Ctx = (__bf16*)(ws + (size_t)28 * 1024 * 1024);

    const dim3 blk(32);
    // Q = x @ Wq^T + bq            -> [B,H,S,DH]
    proj_qkv<S,   false, false><<<(B * S   / 16) * (D / 64), blk, 0, stream>>>(x, mems, Wq, bq, Qb);
    // K = concat(mems,x) @ Wk^T    -> [B,H,SKV,DH]
    proj_qkv<SKV, true,  false><<<(B * SKV / 16) * (D / 64), blk, 0, stream>>>(x, mems, Wk, bk, Kb);
    // V = concat(mems,x) @ Wv^T    -> [B,H,DH,SKV]  (transposed)
    proj_qkv<SKV, true,  true ><<<(B * SKV / 16) * (D / 64), blk, 0, stream>>>(x, mems, Wv, bv, Vt);
    // flash attention (32 query rows / wave) -> Ctx [B,S,D]
    attn_flash<<<B * H * (S / 32), blk, 0, stream>>>(Qb, Kb, Vt, mask, Ctx);
    // out = Ctx @ Wo^T + bo -> d_out [B,S,D] f32
    out_proj<<<(B * S / 16) * (D / 64), blk, 0, stream>>>(Ctx, Wo, bo, out);
}
